// AttentionOutput_1872605741304
// MI455X (gfx1250) — matrix operand order repaired and checked by reading, not compiled, not run on hardware
//
#include <hip/hip_runtime.h>

typedef __attribute__((ext_vector_type(16))) _Float16 v16h;
typedef __attribute__((ext_vector_type(8)))  _Float16 v8h;
typedef __attribute__((ext_vector_type(2)))  _Float16 v2h;
typedef __attribute__((ext_vector_type(8)))  float    v8f;

#define NB 4
#define NN 4096
#define NF 64
#define KP 72   /* sKr/sKi row pitch in halves (144B: conflict-free 16-row phases) */
#define VP 40   /* sVr/sVi (transposed V) row pitch in halves (80B) */
#define WP 72   /* per-wave weight/out scratch row pitch in halves */
#define SCALE (1.0f / 64.0f)   /* 1/sqrt(N) */
#define NEG_SLOPE 0.01f

__device__ __forceinline__ v8f wmma16(v16h a, v16h b, v8f c) {
  // D = A(16x32 f16) * B(32x16 f16) + C(f32)
  return __builtin_amdgcn_wmma_f32_16x16x32_f16(false, a, false, b, (short)0, c,
                                                false, false);
}

// B-fragment / generic: 16 consecutive f16 from LDS (two ds_load_b128)
__device__ __forceinline__ v16h ld16(const _Float16* p) {
  v8h a = *(const v8h*)p;
  v8h b = *(const v8h*)(p + 8);
  v16h r;
#pragma unroll
  for (int i = 0; i < 8; ++i) { r[i] = a[i]; r[i + 8] = b[i]; }
  return r;
}

// A-fragment: halves {base+0..15} of a row (lane offset applied by caller)
__device__ __forceinline__ v16h ldA(const _Float16* p) {
  v8h a = *(const v8h*)p;
  v8h b = *(const v8h*)(p + 8);
  v16h r;
#pragma unroll
  for (int i = 0; i < 8; ++i) { r[i] = a[i]; r[i + 8] = b[i]; }
  return r;
}

__launch_bounds__(128, 1)
__global__ void complex_causal_attn(const float* __restrict__ Qg,
                                    const float* __restrict__ Kg,
                                    const float* __restrict__ Vg,
                                    const float* __restrict__ Wg,
                                    const float* __restrict__ bg,
                                    float* __restrict__ Og) {
  __shared__ __align__(16) _Float16 sKr[32 * KP];
  __shared__ __align__(16) _Float16 sKi[32 * KP];
  __shared__ __align__(16) _Float16 sVr[64 * VP];  // V^T: [f][key]
  __shared__ __align__(16) _Float16 sVi[64 * VP];
  __shared__ __align__(16) _Float16 sW[4 * 16 * WP];  // per-wave 16x64 scratch

  const int tid  = threadIdx.x;
  const int lane = tid & 31;
  const int wave = __builtin_amdgcn_readfirstlane(tid >> 5);  // wave-uniform
  const int b    = blockIdx.y;
  const int qm   = blockIdx.x * 64 + wave * 16;               // wave-uniform
  const size_t bOff = (size_t)b * NN * (NF * 2);

  const int col  = lane & 15;             // N-index inside a 16-wide tile
  const int hsel = (lane >> 4) & 1;       // which half-wave
  const int row0 = hsel * 8;              // C/D row offset for this half
  const int lo   = hsel * 16;             // A-frag lane sub-offset

  // ---- Q A-fragments, resident in VGPRs for the whole kernel ----
  v16h QrA[2], QiA[2], QnA[2];
  {
    const float* qrow = Qg + bOff + (size_t)(qm + col) * (NF * 2);
#pragma unroll
    for (int ff = 0; ff < 2; ++ff) {
#pragma unroll
      for (int hh = 0; hh < 2; ++hh) {
        const int f0 = ff * 32 + lo + hh * 8;   // 8 features, (re,im) interleaved
        const float4* p4 = (const float4*)(qrow + f0 * 2);
#pragma unroll
        for (int c4 = 0; c4 < 4; ++c4) {
          float4 x = p4[c4];
          int e = hh * 8 + c4 * 2;
          QrA[ff][e]     = (_Float16)x.x;
          QiA[ff][e]     = (_Float16)x.y;
          QnA[ff][e]     = (_Float16)(-x.y);
          QrA[ff][e + 1] = (_Float16)x.z;
          QiA[ff][e + 1] = (_Float16)x.w;
          QnA[ff][e + 1] = (_Float16)(-x.w);
        }
      }
    }
  }

  const v8f vzero = {0.f, 0.f, 0.f, 0.f, 0.f, 0.f, 0.f, 0.f};
  v8f outr[4], outi[4];
#pragma unroll
  for (int c = 0; c < 4; ++c) { outr[c] = vzero; outi[c] = vzero; }

  _Float16* wbase = &sW[(wave * 16) * WP];          // this wave's scratch
  const _Float16* wrow = &sW[(wave * 16 + col) * WP];

  const int nkb = blockIdx.x * 2 + 2;               // causal: keys up to block end

  for (int kb = 0; kb < nkb; ++kb) {
    const int kbase = kb * 32;

    // ---- cooperative staging: 32 keys of K (row-major) and V (transposed), f32->f16 ----
#pragma unroll
    for (int it = 0; it < 8; ++it) {
      int idx = tid + it * 128;        // 0..1023 float4 slots
      int key = idx >> 5;              // 0..31
      int q4  = idx & 31;              // float4 within the 128-f32 row
      int f0  = q4 * 2;                // feature pair index
      const float* kr = Kg + bOff + (size_t)(kbase + key) * (NF * 2);
      float4 x = *(const float4*)(kr + q4 * 4);     // (Kr[f0],Ki[f0],Kr[f0+1],Ki[f0+1])
      v2h tr; tr[0] = (_Float16)x.x; tr[1] = (_Float16)x.z;
      v2h ti; ti[0] = (_Float16)x.y; ti[1] = (_Float16)x.w;
      *(v2h*)&sKr[key * KP + f0] = tr;
      *(v2h*)&sKi[key * KP + f0] = ti;
      const float* vr = Vg + bOff + (size_t)(kbase + key) * (NF * 2);
      float4 y = *(const float4*)(vr + q4 * 4);
      sVr[f0 * VP + key]       = (_Float16)y.x;
      sVr[(f0 + 1) * VP + key] = (_Float16)y.z;
      sVi[f0 * VP + key]       = (_Float16)y.y;
      sVi[(f0 + 1) * VP + key] = (_Float16)y.w;
    }
    __syncthreads();

    if (kbase <= qm + 15) {   // wave-uniform -> scalar branch, EXEC stays all-ones
      // ---- scores for the two 16-key halves of this block ----
#pragma unroll
      for (int h = 0; h < 2; ++h) {
        const int kh = kbase + h * 16;
        const _Float16* krow = &sKr[(h * 16 + col) * KP];
        const _Float16* irow = &sKi[(h * 16 + col) * KP];
        const int sh = hsel * 16;
        v16h Kr0 = ld16(krow + sh);
        v16h Kr1 = ld16(krow + 32 + sh);
        v16h Ki0 = ld16(irow + sh);
        v16h Ki1 = ld16(irow + 32 + sh);

        v8f sr = vzero, si = vzero;
        sr = wmma16(QrA[0], Kr0, sr);   // Qr*Kr^T
        sr = wmma16(QrA[1], Kr1, sr);
        sr = wmma16(QnA[0], Ki0, sr);   // -Qi*Ki^T
        sr = wmma16(QnA[1], Ki1, sr);
        si = wmma16(QrA[0], Ki0, si);   // Qr*Ki^T
        si = wmma16(QrA[1], Ki1, si);
        si = wmma16(QiA[0], Kr0, si);   // Qi*Kr^T
        si = wmma16(QiA[1], Kr1, si);

        const bool needMask = (kh + 15 > qm);   // wave-uniform
#pragma unroll
        for (int r = 0; r < 8; ++r) {
          float fa = sr[r] * SCALE; fa = (fa >= 0.f) ? fa : fa * NEG_SLOPE;
          float fb = si[r] * SCALE; fb = (fb >= 0.f) ? fb : fb * NEG_SLOPE;
          if (needMask && (kh + col > qm + row0 + r)) { fa = 0.f; fb = 0.f; }
          // Wr tile in cols 0..31, Wi tile in cols 32..63 of the wave scratch
          wbase[(row0 + r) * WP + h * 16 + col]      = (_Float16)fa;
          wbase[(row0 + r) * WP + 32 + h * 16 + col] = (_Float16)fb;
        }
      }

      // ---- (weights 16x32) @ (V 32x64), accumulate into out tiles ----
      v16h WrA = ldA(wrow + lo);        // A-frag of Wr
      v16h WiA = ldA(wrow + 32 + lo);   // A-frag of Wi
#pragma unroll
      for (int c = 0; c < 4; ++c) {
        v16h Vr = ld16(&sVr[(c * 16 + col) * VP + hsel * 16]);
        v16h Vi = ld16(&sVi[(c * 16 + col) * VP + hsel * 16]);
        outr[c] = wmma16(WrA, Vr, outr[c]);
        outi[c] = wmma16(WiA, Vi, outi[c]);
      }
    }
    __syncthreads();
  }

  // ---- epilogue: y = out @ W_att^T + b_att (contraction over g=64) ----
  v16h WB[4][2];
#pragma unroll
  for (int c = 0; c < 4; ++c) {
    const int f = c * 16 + col;
#pragma unroll
    for (int ff = 0; ff < 2; ++ff) {
      const int g0 = ff * 32 + hsel * 16;
      const float4* p4 = (const float4*)(Wg + f * 64 + g0);
      v16h t;
#pragma unroll
      for (int c4 = 0; c4 < 4; ++c4) {
        float4 x = p4[c4];
        t[c4 * 4 + 0] = (_Float16)x.x;
        t[c4 * 4 + 1] = (_Float16)x.y;
        t[c4 * 4 + 2] = (_Float16)x.z;
        t[c4 * 4 + 3] = (_Float16)x.w;
      }
      WB[c][ff] = t;
    }
  }

  v8f yr[4], yi[4];
  // real component
#pragma unroll
  for (int c = 0; c < 4; ++c)
#pragma unroll
    for (int r = 0; r < 8; ++r)
      wbase[(row0 + r) * WP + c * 16 + col] = (_Float16)outr[c][r];
  {
    v16h OA0 = ldA(wrow + lo);
    v16h OA1 = ldA(wrow + 32 + lo);
#pragma unroll
    for (int c = 0; c < 4; ++c) {
      v8f y = vzero;
      y = wmma16(OA0, WB[c][0], y);
      y = wmma16(OA1, WB[c][1], y);
      yr[c] = y;
    }
  }
  // imaginary component
#pragma unroll
  for (int c = 0; c < 4; ++c)
#pragma unroll
    for (int r = 0; r < 8; ++r)
      wbase[(row0 + r) * WP + c * 16 + col] = (_Float16)outi[c][r];
  {
    v16h OA0 = ldA(wrow + lo);
    v16h OA1 = ldA(wrow + 32 + lo);
#pragma unroll
    for (int c = 0; c < 4; ++c) {
      v8f y = vzero;
      y = wmma16(OA0, WB[c][0], y);
      y = wmma16(OA1, WB[c][1], y);
      yi[c] = y;
    }
  }

  // bias + interleaved (re,im) store
#pragma unroll
  for (int c = 0; c < 4; ++c) {
    const int f = c * 16 + col;
    const float bv = bg[f];
#pragma unroll
    for (int r = 0; r < 8; ++r) {
      const size_t i = (size_t)(qm + row0 + r);
      float2 o;
      o.x = yr[c][r] + bv;
      o.y = yi[c][r] + bv;
      *(float2*)(Og + bOff + i * (NF * 2) + (size_t)f * 2) = o;
    }
  }
}

extern "C" void kernel_launch(void* const* d_in, const int* in_sizes, int n_in,
                              void* d_out, int out_size, void* d_ws, size_t ws_size,
                              hipStream_t stream) {
  (void)in_sizes; (void)n_in; (void)out_size; (void)d_ws; (void)ws_size;
  const float* Q  = (const float*)d_in[0];
  const float* K  = (const float*)d_in[1];
  const float* V  = (const float*)d_in[2];
  const float* Wa = (const float*)d_in[3];
  const float* ba = (const float*)d_in[4];
  float* out = (float*)d_out;

  dim3 grid(NN / 64, NB);   // 64 query-blocks of 64 rows x 4 batches = 256 WGs
  dim3 block(128);          // 4 wave32s, one 16-row query tile each
  hipLaunchKernelGGL(complex_causal_attn, grid, block, 0, stream,
                     Q, K, V, Wa, ba, out);
}
